// StateLSTM_47596827574783
// MI455X (gfx1250) — matrix operand order repaired and checked by reading, not compiled, not run on hardware
//
#include <hip/hip_runtime.h>
#include <hip/hip_bf16.h>

typedef __attribute__((ext_vector_type(16))) _Float16 v16h;
typedef __attribute__((ext_vector_type(2)))  __fp16   v2fp16;
typedef __attribute__((ext_vector_type(8)))  float    v8f;

#define BS   256
#define J    50
#define O    50
#define NSEQ (BS*J)          // 12800 sequences
#define EH   64              // LSTM hidden
#define GATES 256            // 4*EH
#define DSET 192             // Set2Set dim 3*E
#define STEPS 51

// workspace layout (bytes)
#define WS_TILES_OFF 0        // 48KB: 32 Whh A-tiles (32KB) + 16 U A-tiles (16KB)
#define WS_H2_OFF    49152    // 192 floats (Set2Set constant hidden)
#define WS_STATE_OFF 65536    // 12800*192 floats ("state" = [je | dyn])

union HV { v16h v; int4 i4[2]; unsigned d[8]; };

// ---- fast transcendentals: prefer native v_tanh_f32 (CDNA5 trans op) -------
__device__ __forceinline__ float frcp_fast(float x){
#if __has_builtin(__builtin_amdgcn_rcpf)
  return __builtin_amdgcn_rcpf(x);
#else
  return 1.0f / x;
#endif
}
__device__ __forceinline__ float ftanh(float x){
#if __has_builtin(__builtin_amdgcn_tanhf)
  return __builtin_amdgcn_tanhf(x);
#else
  x = fminf(10.f, fmaxf(-10.f, x));
  float e = __expf(2.f*x);
  return (e-1.f) * frcp_fast(e+1.f);
#endif
}
__device__ __forceinline__ float fsigm(float x){
#if __has_builtin(__builtin_amdgcn_tanhf)
  return fmaf(0.5f, __builtin_amdgcn_tanhf(0.5f*x), 0.5f);   // 1 trans + 1 mul + 1 fma
#else
  return frcp_fast(1.0f + __expf(-x));
#endif
}
__device__ __forceinline__ unsigned pk2(float a, float b){
#if __has_builtin(__builtin_amdgcn_cvt_pkrtz)
  union{ v2fp16 h; unsigned u; } t; t.h = __builtin_amdgcn_cvt_pkrtz(a, b); return t.u;
#else
  union{ _Float16 h[2]; unsigned u; } t;
  t.h[0]=(_Float16)a; t.h[1]=(_Float16)b; return t.u;
#endif
}

// ---------------------------------------------------------------------------
// K0: fold input projection to rank-3 (u1,u2,u0), swizzle Whh & U into the
// exact 16-bit WMMA A-matrix VGPR/lane layout, and compute the Set2Set
// bias-only hidden vector h2.
// ---------------------------------------------------------------------------
__global__ void k0_precompute(const float* __restrict__ Wmac, const float* __restrict__ bmac,
                              const float* __restrict__ Wjt,  const float* __restrict__ bjt,
                              const float* __restrict__ Wih,  const float* __restrict__ Whh,
                              const float* __restrict__ bih,  const float* __restrict__ bhh,
                              const float* __restrict__ bih2, const float* __restrict__ bhh2,
                              unsigned* __restrict__ wsTiles, float* __restrict__ wsH2) {
  __shared__ float su1[GATES], su2[GATES], su0[GATES];
  int tid = threadIdx.x;
  {
    int row = tid;                       // 256 threads == 256 gate rows
    float a = 0.f, b = 0.f, c0 = 0.f;
    for (int e = 0; e < EH; e++) {
      float w0 = Wih[row*128 + e];
      float w1 = Wih[row*128 + 64 + e];
      a  += w0 * Wmac[e];
      b  += w1 * Wjt[e];
      c0 += w0 * bmac[e] + w1 * bjt[e];
    }
    su1[row] = a; su2[row] = b; su0[row] = c0 + bih[row] + bhh[row];
  }
  __syncthreads();
  // Whh A-tiles: tile (mt,kh) covers rows mt*16..+15, K = kh*32..+31.
  // 16-bit A layout: lane m (0-15) kbase=0, lane m+16 kbase=8;
  //   dword v in 0..3 -> K = kbase+2v,+1 ; v in 4..7 -> K = 16+kbase+2(v-4),+1
  for (int i = tid; i < 32*256; i += blockDim.x) {
    int tileIdx = i >> 8, w = i & 255;
    int lane = w >> 3, v = w & 7;
    int mt = tileIdx >> 1, kh = tileIdx & 1;
    int m = lane & 15, kb = (lane >> 4) * 8;
    int K = (v < 4) ? (kb + 2*v) : (16 + kb + 2*(v-4));
    int row = mt*16 + m;
    int k0 = kh*32 + K;
    wsTiles[i] = pk2(Whh[row*64 + k0], Whh[row*64 + k0 + 1]);
  }
  // U A-tiles (K0=u1, K1=u2, K2=u0, rest zero)
  for (int i = tid; i < 16*256; i += blockDim.x) {
    int mt = i >> 8, w = i & 255;
    int lane = w >> 3, v = w & 7;
    int m = lane & 15, hi = lane >> 4;
    int row = mt*16 + m;
    unsigned dd = 0u;
    if (hi == 0 && v == 0) dd = pk2(su1[row], su2[row]);
    else if (hi == 0 && v == 1) dd = pk2(su0[row], 0.f);
    wsTiles[8192 + i] = dd;
  }
  // Set2Set first cell from zero state: gates are pure biases
  if (tid < DSET) {
    float gi = bih2[tid]          + bhh2[tid];
    float gg = bih2[2*DSET+tid]   + bhh2[2*DSET+tid];
    float go = bih2[3*DSET+tid]   + bhh2[3*DSET+tid];
    float c  = fsigm(gi) * ftanh(gg);          // f-gate multiplies c==0
    wsH2[tid] = fsigm(go) * ftanh(c);
  }
}

// ---------------------------------------------------------------------------
// K1: batched 51-step LSTM via WMMA (transposed gates).  One wave = 16 seqs.
//   gates^T[row,seq] = sum_k Whh[row,k] * h[seq,k]  +  u1*prec + u2*jt + u0
// B-matrix (h^T / scalar matrix) layout: lane%16 = seq(col), K = 8*hi+0..7, 16+8*hi+0..7 per
// lane -> h lives contiguously per-seq in LDS (1x ds_store_b128 per chunk).
// ---------------------------------------------------------------------------
__global__ void __launch_bounds__(128)
k1_lstm(const float* __restrict__ jt, const int* __restrict__ prec,
        const int* __restrict__ jstate, const unsigned* __restrict__ wsTiles,
        float* __restrict__ state) {
  extern __shared__ __align__(16) char smem[];
  unsigned* tilesS = (unsigned*)smem;                       // 12288 dwords (48KB)
  int tid = threadIdx.x, lane = tid & 31, wave = tid >> 5;

  { // cooperative copy of pre-swizzled A-tiles into LDS
    const int4* src = (const int4*)wsTiles;
    int4* dst = (int4*)tilesS;
    for (int i = tid; i < 3072; i += blockDim.x) dst[i] = src[i];
  }
  unsigned* hbuf  = (unsigned*)(smem + 49152 + wave*5376);  // 512 dwords: h f16 [16 seq][64]
  unsigned* inbuf = hbuf + 512;                             // 832 dwords: packed (prec,jt) f16x2
  for (int i = lane; i < 512; i += 32) hbuf[i] = 0u;

  int seql = lane & 15, hi = lane >> 4;
  int tile = blockIdx.x * 4 + wave;
  int seqg = tile * 16 + seql;

  // stage per-step scalars (reversed sequence; s=0 is the pad column)
  for (int s = hi*26; s < (hi ? STEPS : 26); ++s) {
    int idx = 50 - s;
    int p; float t;
    if (idx < 50) { p = prec[seqg*50 + idx]; t = jt[seqg*50 + idx]; }
    else          { p = 50;                  t = 0.f; }
    inbuf[seql*52 + s] = pk2((float)p, t);
  }
  int srec = 50 - jstate[seqg];
  __syncthreads();

  float c_[4][8], je[4][8];
  #pragma unroll
  for (int t = 0; t < 4; t++)
    #pragma unroll
    for (int r = 0; r < 8; r++) { c_[t][r] = 0.f; je[t][r] = 0.f; }

  const int4* hb4 = (const int4*)hbuf;
  const int4* t4  = (const int4*)tilesS;

  for (int s = 0; s < STEPS; s++) {
    // B-matrix of the rank-3 input term: K0=prec, K1=jt, K2=1 (lanes>=16: K8..15, K24..31 = 0)
    HV bs;
    bs.d[0] = (lane < 16) ? inbuf[seql*52 + s] : 0u;
    bs.d[1] = (lane < 16) ? 0x00003C00u : 0u;   // (1.0h, 0h)
    #pragma unroll
    for (int q = 2; q < 8; q++) bs.d[q] = 0u;
    // B-matrix of h^T, K-halves 0..31 / 32..63 (per-lane K = 8*hi+0..7, 16+8*hi+0..7 from hbuf)
    HV bh0, bh1;
    bh0.i4[0] = hb4[seql*8 + 0 + hi];
    bh0.i4[1] = hb4[seql*8 + 0 + hi + 2];
    bh1.i4[0] = hb4[seql*8 + 4 + hi];
    bh1.i4[1] = hb4[seql*8 + 4 + hi + 2];

    bool rec = (s == srec);
    #pragma unroll
    for (int t = 0; t < 4; t++) {                 // hidden-chunk t: elems t*16..+15
      v8f acc[4];
      #pragma unroll
      for (int g = 0; g < 4; g++) {               // gates i,f,g,o
        int mt = g*4 + t;                         // row-tile (16 gate rows)
        HV au, a0, a1;
        au.i4[0] = t4[2048 + mt*64 + lane*2];
        au.i4[1] = t4[2048 + mt*64 + lane*2 + 1];
        a0.i4[0] = t4[(mt*2 + 0)*64 + lane*2];
        a0.i4[1] = t4[(mt*2 + 0)*64 + lane*2 + 1];
        a1.i4[0] = t4[(mt*2 + 1)*64 + lane*2];
        a1.i4[1] = t4[(mt*2 + 1)*64 + lane*2 + 1];
        v8f z = {0.f,0.f,0.f,0.f,0.f,0.f,0.f,0.f};
        v8f a = __builtin_amdgcn_wmma_f32_16x16x32_f16(false, au.v, false, bs.v,  (short)0, z, false, false);
        a     = __builtin_amdgcn_wmma_f32_16x16x32_f16(false, a0.v, false, bh0.v, (short)0, a, false, false);
        a     = __builtin_amdgcn_wmma_f32_16x16x32_f16(false, a1.v, false, bh1.v, (short)0, a, false, false);
        acc[g] = a;
      }
      // elementwise LSTM cell update for 8 elems (this lane) of this chunk
      float hn[8];
      #pragma unroll
      for (int r = 0; r < 8; r++) {
        float cn = fsigm(acc[1][r]) * c_[t][r] + fsigm(acc[0][r]) * ftanh(acc[2][r]);
        c_[t][r] = cn;
        float hv = fsigm(acc[3][r]) * ftanh(cn);
        hn[r] = hv;
        if (rec) je[t][r] = hv;                   // capture h at t = 50 - job_state
      }
      int4 hp;
      hp.x = (int)pk2(hn[0], hn[1]); hp.y = (int)pk2(hn[2], hn[3]);
      hp.z = (int)pk2(hn[4], hn[5]); hp.w = (int)pk2(hn[6], hn[7]);
      ((int4*)hbuf)[seql*8 + t*2 + hi] = hp;      // 16B contiguous per-seq store
    }
  }
  // state[:, 0:64] = je
  float* rowp = state + (size_t)seqg * DSET + hi*8;
  #pragma unroll
  for (int t = 0; t < 4; t++)
    #pragma unroll
    for (int r = 0; r < 8; r++)
      rowp[t*16 + r] = je[t][r];
}

// ---------------------------------------------------------------------------
// K2a: dynamic features (machine-load scatter via LDS f32 atomics) + 7->128
// projection into state[:, 64:192].
// ---------------------------------------------------------------------------
__global__ void k2_dyn(const float* __restrict__ jt, const int* __restrict__ prec,
                       const int* __restrict__ jstate, const float* __restrict__ jest,
                       const float* __restrict__ mutil, const float* __restrict__ mk,
                       const float* __restrict__ Wdyn, const float* __restrict__ bdyn,
                       float* __restrict__ state) {
  __shared__ float twm[51];
  __shared__ float feats[J][7];
  int b = blockIdx.x, tid = threadIdx.x;
  if (tid < 51) twm[tid] = 0.f;
  __syncthreads();
  for (int i = tid; i < J*O; i += blockDim.x) {
    int m = prec[b*J*O + i];                        // permutation: m in [0,50)
    atomicAdd(&twm[m], jt[b*J*O + i]);
  }
  __syncthreads();
  if (tid < J) {
    int j = tid;
    int st = jstate[b*J + j];
    const float* row = jt + (size_t)(b*J + j) * O;
    float jpt = (st < O) ? row[st] : 0.f;
    float jst_ = jest[b*J + j];
    float twr = 0.f;
    for (int o = st; o < O; o++) twr += row[o];
    int mac = (st < O) ? prec[(b*J + j)*O + st] : 50;
    float mu = (mac < 50) ? mutil[b*50 + mac] : 0.f;
    float trm = twm[mac] - mu;                      // twm[50] == 0
    feats[j][0]=jpt; feats[j][1]=jst_; feats[j][2]=jst_+jpt; feats[j][3]=twr;
    feats[j][4]=mu;  feats[j][5]=trm;  feats[j][6]=mk[b];
  }
  __syncthreads();
  for (int i = tid; i < J*128; i += blockDim.x) {
    int j = i >> 7, d = i & 127;
    float a = bdyn[d];
    #pragma unroll
    for (int k = 0; k < 7; k++) a += feats[j][k] * Wdyn[d*7 + k];
    state[(size_t)(b*J + j)*DSET + 64 + d] = a;
  }
}

// ---------------------------------------------------------------------------
// K2b: Set2Set attention (h2 is the bias-only constant hidden), out = [h2 | r]
// ---------------------------------------------------------------------------
__global__ void k2_s2s(const float* __restrict__ state, const float* __restrict__ wsH2,
                       float* __restrict__ out) {
  __shared__ float h2s[DSET];
  __shared__ float ev[J];
  __shared__ float ew[J];
  int b = blockIdx.x, tid = threadIdx.x;
  if (tid < DSET) h2s[tid] = wsH2[tid];
  __syncthreads();
  if (tid < J) {
    const float* row = state + (size_t)(b*J + tid) * DSET;
    float d = 0.f;
    for (int k = 0; k < DSET; k++) d += row[k] * h2s[k];
    ev[tid] = d;
  }
  __syncthreads();
  float mx = -1e30f;
  for (int j = 0; j < J; j++) mx = fmaxf(mx, ev[j]);
  if (tid < J) ew[tid] = __expf(ev[tid] - mx);
  __syncthreads();
  float sum = 0.f;
  for (int j = 0; j < J; j++) sum += ew[j];
  float inv = 1.f / sum;
  if (tid < DSET) {
    float rd = 0.f;
    for (int j = 0; j < J; j++) rd += ew[j] * state[(size_t)(b*J + j)*DSET + tid];
    out[b*2*DSET + tid] = h2s[tid];
    out[b*2*DSET + DSET + tid] = rd * inv;
  }
}

extern "C" void kernel_launch(void* const* d_in, const int* in_sizes, int n_in,
                              void* d_out, int out_size, void* d_ws, size_t ws_size,
                              hipStream_t stream) {
  const float* jt    = (const float*)d_in[0];
  const int*   prec  = (const int*)d_in[1];
  const int*   jstate= (const int*)d_in[2];
  const float* jest  = (const float*)d_in[3];
  const float* mutil = (const float*)d_in[4];
  const float* mk    = (const float*)d_in[5];
  const float* Wmac  = (const float*)d_in[6];
  const float* bmac  = (const float*)d_in[7];
  const float* Wjt   = (const float*)d_in[8];
  const float* bjt   = (const float*)d_in[9];
  const float* Wih   = (const float*)d_in[10];
  const float* Whh   = (const float*)d_in[11];
  const float* bih   = (const float*)d_in[12];
  const float* bhh   = (const float*)d_in[13];
  const float* Wdyn  = (const float*)d_in[14];
  const float* bdyn  = (const float*)d_in[15];
  // d_in[16] (Wih2) and d_in[17] (Whh2) are mathematically dead: the Set2Set
  // cell runs once from q_star=0, h=0 -> gates are biases only.
  const float* bih2  = (const float*)d_in[18];
  const float* bhh2  = (const float*)d_in[19];

  unsigned* wsTiles = (unsigned*)d_ws;
  float* wsH2  = (float*)((char*)d_ws + WS_H2_OFF);
  float* state = (float*)((char*)d_ws + WS_STATE_OFF);
  float* out   = (float*)d_out;

  k0_precompute<<<1, 256, 0, stream>>>(Wmac, bmac, Wjt, bjt, Wih, Whh, bih, bhh,
                                       bih2, bhh2, wsTiles, wsH2);
  // 12800 seqs / (16 per wave * 4 waves) = 200 blocks; 70656B dynamic LDS
  k1_lstm<<<200, 128, 70656, stream>>>(jt, prec, jstate, wsTiles, state);
  k2_dyn<<<BS, 256, 0, stream>>>(jt, prec, jstate, jest, mutil, mk, Wdyn, bdyn, state);
  k2_s2s<<<BS, 256, 0, stream>>>(state, wsH2, out);
}
